// EEMoE_40364102648322
// MI455X (gfx1250) — compile-verified, matrix operand order checked by reading, not run on hardware
//
#include <hip/hip_runtime.h>
#include <hip/hip_bf16.h>
#include <math.h>

typedef __attribute__((ext_vector_type(16))) _Float16 v16h;
typedef __attribute__((ext_vector_type(8)))  _Float16 v8h;
typedef __attribute__((ext_vector_type(8)))  float    v8f;
typedef __attribute__((ext_vector_type(4)))  unsigned int u32x4;
typedef __attribute__((ext_vector_type(8)))  int      i32x8;
typedef __attribute__((ext_vector_type(4)))  int      i32x4;

#define C_DIM 256
#define E_NUM 5
#define HID_D 1024
#define H_IMG 64
#define W_IMG 64
#define NTOK  8192
#define KCONV 2304   // 9 * 256

#if __has_include(<hip/amd_detail/amd_gfx1250_TDM.h>)
#define TDM_6ARG 1   // amdgpu-toolchain / therock-10.0 headers -> 6-arg builtin
#else
#define TDM_6ARG 0   // ROCm 7.2 -> 5-arg builtin
#endif

// ---------------------------------------------------------------------------
// TDM: 1-D contiguous tile load Global -> LDS (n_qwords x 8B), one descriptor.
// D# group0: count=1, lds_addr, global_addr[56:0], type=2.
// D# group1: data_size=8B, tensor_dim0=tile_dim0=n_qwords, tensor_dim1=1,
//            tile_dim1=1, stride=n_qwords; no multicast/pad/iterate.
// ---------------------------------------------------------------------------
__device__ __forceinline__ void tdm_load_1d(unsigned lds_off, const void* gptr,
                                            unsigned n_qwords) {
  unsigned long long ga = (unsigned long long)gptr;
  u32x4 g0;
  g0[0] = 1u;                                        // count=1, user mode
  g0[1] = lds_off;                                   // LDS byte address
  g0[2] = (unsigned)(ga & 0xffffffffu);              // global_addr[31:0]
  g0[3] = (unsigned)((ga >> 32) & 0x01ffffffu)       // global_addr[56:32]
        | (2u << 30);                                // type = 2 ("image")
  i32x8 g1;
  g1[0] = (int)(3u << 16);                           // data_size = 3 (8 bytes)
  g1[1] = (int)((n_qwords & 0xffffu) << 16);         // tensor_dim0[15:0]
  g1[2] = (int)((n_qwords >> 16) & 0xffffu)          // tensor_dim0[31:16]
        | (1 << 16);                                 // tensor_dim1 = 1
  g1[3] = (int)((n_qwords & 0xffffu) << 16);         // tile_dim0
  g1[4] = 1;                                         // tile_dim1=1, tile_dim2=0
  g1[5] = (int)n_qwords;                             // tensor_dim0_stride lo
  g1[6] = 0;
  g1[7] = 0;
  i32x4 z4 = {0, 0, 0, 0};
#if TDM_6ARG
  i32x8 z8 = {0, 0, 0, 0, 0, 0, 0, 0};
  __builtin_amdgcn_tensor_load_to_lds(g0, g1, z4, z4, z8, 0);
#else
  __builtin_amdgcn_tensor_load_to_lds(g0, g1, z4, z4, 0);
#endif
}

// Per-lane async copy of one 16B chunk Global -> LDS (ASYNCcnt-tracked).
__device__ __forceinline__ void async_copy_b128(unsigned lds_off, const void* gptr) {
  asm volatile("global_load_async_to_lds_b128 %0, %1, off"
               :: "v"(lds_off), "v"(gptr) : "memory");
}
__device__ __forceinline__ void wait_asynccnt0() {
  asm volatile("s_wait_asynccnt 0x0" ::: "memory");
}

// ---------------------------------------------------------------------------
// Fragment loader: row-major [16][stride] f16 tile -> v16h WMMA fragment.
// Per ISA 16-bit A/B layout: row = lane%16; K chunks at 8*(lane/16) and +16.
// ---------------------------------------------------------------------------
__device__ __forceinline__ v16h frag_load(const _Float16* base, int stride, int lane) {
  const int r  = lane & 15;
  const int k0 = (lane >> 4) << 3;     // 0 or 8
  const _Float16* p = base + r * stride + k0;
  v8h lo = *(const v8h*)(p);
  v8h hi = *(const v8h*)(p + 16);
  v16h f;
#pragma unroll
  for (int i = 0; i < 8; ++i) { f[i] = lo[i]; f[i + 8] = hi[i]; }
  return f;
}

__device__ __forceinline__ v8f wmma16(v16h a, v16h b, v8f c) {
  return __builtin_amdgcn_wmma_f32_16x16x32_f16(false, a, false, b, (short)0, c,
                                                false, false);
}

// ---------------------------------------------------------------------------
// K0: DEA-Net weight re-parameterization -> fused 3x3 weight, transposed f16
//     Wt[co][k] with k = (ky*3+kx)*256 + ci; fused bias bf[256].
// ---------------------------------------------------------------------------
__global__ void k_reparam(const float* __restrict__ w_cd, const float* __restrict__ b_cd,
                          const float* __restrict__ w_hd, const float* __restrict__ b_hd,
                          const float* __restrict__ w_vd, const float* __restrict__ b_vd,
                          const float* __restrict__ w_ad, const float* __restrict__ b_ad,
                          const float* __restrict__ w_std, const float* __restrict__ b_std,
                          _Float16* __restrict__ Wt, float* __restrict__ bf) {
  int gid = blockIdx.x * blockDim.x + threadIdx.x;
  if (gid < C_DIM) {
    bf[gid] = b_cd[gid] + b_hd[gid] + b_vd[gid] + b_ad[gid] + b_std[gid];
  }
  if (gid >= C_DIM * C_DIM * 9) return;
  int co  = gid / (C_DIM * 9);
  int rem = gid % (C_DIM * 9);
  int ci  = rem / 9;
  int p   = rem % 9;
  const int base9 = (co * C_DIM + ci) * 9;
  const int base3 = (co * C_DIM + ci) * 3;

  float wc = w_cd[base9 + p];
  if (p == 4) {
    float s = 0.f;
#pragma unroll
    for (int q = 0; q < 9; ++q) s += w_cd[base9 + q];
    wc -= s;
  }
  float whv = 0.f;
  {
    int col = p % 3, row = p / 3;
    if (col == 0) whv += w_hd[base3 + row];
    else if (col == 2) whv -= w_hd[base3 + row];
    if (p < 3) whv += w_vd[base3 + p];
    else if (p >= 6) whv -= w_vd[base3 + (p - 6)];
  }
  const int perm[9] = {3, 0, 1, 6, 4, 2, 7, 8, 5};
  float wa = w_ad[base9 + p] - w_ad[base9 + perm[p]];
  float w = wc + whv + wa + w_std[base9 + p];
  Wt[(size_t)co * KCONV + p * C_DIM + ci] = (_Float16)w;
}

// ---------------------------------------------------------------------------
// K0b: transpose + f16-convert expert weights.
// ---------------------------------------------------------------------------
__global__ void k_transpose_experts(const float* __restrict__ We1,
                                    const float* __restrict__ We2,
                                    _Float16* __restrict__ Wt1,
                                    _Float16* __restrict__ Wt2) {
  int gid = blockIdx.x * blockDim.x + threadIdx.x;
  if (gid >= E_NUM * C_DIM * HID_D) return;
  int e   = gid / (C_DIM * HID_D);
  int rem = gid % (C_DIM * HID_D);
  const size_t eb = (size_t)e * C_DIM * HID_D;
  {
    int n = rem % HID_D;
    int k = rem / HID_D;
    Wt1[eb + (size_t)n * C_DIM + k] = (_Float16)We1[eb + (size_t)k * HID_D + n];
  }
  {
    int n = rem % C_DIM;
    int k = rem / C_DIM;
    Wt2[eb + (size_t)n * HID_D + k] = (_Float16)We2[eb + (size_t)k * C_DIM + n];
  }
}

// ---------------------------------------------------------------------------
// K1: 3x3 conv as implicit GEMM. M=8192 tokens, N=256, K=2304.
// ---------------------------------------------------------------------------
__global__ void __launch_bounds__(256)
k_conv(const float* __restrict__ x, const _Float16* __restrict__ Wt,
       const float* __restrict__ bf, float* __restrict__ t,
       _Float16* __restrict__ tf16) {
  __shared__ _Float16 As[16 * 256];
  const int tid  = threadIdx.x;
  const int lane = tid & 31;
  const int wv   = tid >> 5;
  const int m0   = blockIdx.x * 16;
  const int b    = m0 >> 12;
  const int h    = (m0 >> 6) & 63;
  const int w0   = m0 & 63;
  const int nt0  = wv * 2;

  v8f acc0, acc1;
#pragma unroll
  for (int i = 0; i < 8; ++i) { acc0[i] = 0.f; acc1[i] = 0.f; }

  for (int kyx = 0; kyx < 9; ++kyx) {
    const int ky = kyx / 3, kx = kyx % 3;
    const int hh = h + ky - 1;
    // prefetch next slab's B rows for this wave's two N-tiles
    if (kyx < 8) {
      __builtin_prefetch(Wt + (size_t)(nt0 + 0) * 16 * KCONV + (kyx + 1) * 256 + lane * 8, 0, 0);
      __builtin_prefetch(Wt + (size_t)(nt0 + 1) * 16 * KCONV + (kyx + 1) * 256 + lane * 8, 0, 0);
    }
    __syncthreads();
    // cooperative im2col slab fill: 16 rows x 256 ci (f16) with halo zeroing
#pragma unroll
    for (int j = 0; j < 2; ++j) {
      int c  = tid + 256 * j;
      int r  = c >> 5;
      int cc = (c & 31) << 3;
      int ww = w0 + r + kx - 1;
      v8h hv;
      if (hh >= 0 && hh < H_IMG && ww >= 0 && ww < W_IMG) {
        const float4* xp =
            (const float4*)(x + (((size_t)(b * H_IMG + hh) * W_IMG + ww) * C_DIM + cc));
        float4 u0 = xp[0], u1 = xp[1];
        hv[0] = (_Float16)u0.x; hv[1] = (_Float16)u0.y;
        hv[2] = (_Float16)u0.z; hv[3] = (_Float16)u0.w;
        hv[4] = (_Float16)u1.x; hv[5] = (_Float16)u1.y;
        hv[6] = (_Float16)u1.z; hv[7] = (_Float16)u1.w;
      } else {
#pragma unroll
        for (int i = 0; i < 8; ++i) hv[i] = (_Float16)0.f;
      }
      *(v8h*)(&As[r * 256 + cc]) = hv;
    }
    __syncthreads();
#pragma unroll
    for (int kb = 0; kb < 8; ++kb) {
      v16h a  = frag_load(&As[kb * 32], 256, lane);
      v16h b0 = frag_load(Wt + (size_t)(nt0 + 0) * 16 * KCONV + kyx * 256 + kb * 32,
                          KCONV, lane);
      v16h b1 = frag_load(Wt + (size_t)(nt0 + 1) * 16 * KCONV + kyx * 256 + kb * 32,
                          KCONV, lane);
      acc0 = wmma16(a, b0, acc0);
      acc1 = wmma16(a, b1, acc1);
    }
  }
#pragma unroll
  for (int i = 0; i < 8; ++i) {
    int m     = i + ((lane >> 4) << 3);
    int token = m0 + m;
    {
      int ch  = nt0 * 16 + (lane & 15);
      float v = acc0[i] + bf[ch];
      t[(size_t)token * C_DIM + ch]    = v;
      tf16[(size_t)token * C_DIM + ch] = (_Float16)v;
    }
    {
      int ch  = (nt0 + 1) * 16 + (lane & 15);
      float v = acc1[i] + bf[ch];
      t[(size_t)token * C_DIM + ch]    = v;
      tf16[(size_t)token * C_DIM + ch] = (_Float16)v;
    }
  }
}

// ---------------------------------------------------------------------------
// K2: noisy top-1 router. One thread per token.
// ---------------------------------------------------------------------------
__global__ void k_router(const float* __restrict__ t,
                         const float* __restrict__ Wg, const float* __restrict__ bg,
                         const float* __restrict__ Wn, const float* __restrict__ bn,
                         const float* __restrict__ noise, int* __restrict__ idx) {
  int token = blockIdx.x * blockDim.x + threadIdx.x;
  if (token >= NTOK) return;
  float lg[E_NUM], nl[E_NUM];
#pragma unroll
  for (int e = 0; e < E_NUM; ++e) { lg[e] = bg[e]; nl[e] = bn[e]; }
  const float* tr = t + (size_t)token * C_DIM;
  for (int c = 0; c < C_DIM; ++c) {
    float tv = tr[c];
#pragma unroll
    for (int e = 0; e < E_NUM; ++e) {
      lg[e] += tv * Wg[c * E_NUM + e];
      nl[e] += tv * Wn[c * E_NUM + e];
    }
  }
  int best = 0; float bv = -3.4e38f;
#pragma unroll
  for (int e = 0; e < E_NUM; ++e) {
    float sp = (nl[e] > 30.f) ? nl[e] : log1pf(expf(nl[e]));
    float nz = lg[e] + noise[(size_t)token * E_NUM + e] * sp;
    if (nz > bv) { bv = nz; best = e; }
  }
  idx[token] = best;
}

// ---------------------------------------------------------------------------
// K3: expert GEMM1: h = relu(tf16 @ We1[e] + be1[e]).  M=8192,N=1024,K=256.
// A tile (8KB contiguous) staged via global_load_async_to_lds_b128.
// ---------------------------------------------------------------------------
__global__ void __launch_bounds__(128)
k_moe1(int e, const _Float16* __restrict__ tf16, const _Float16* __restrict__ Wt1,
       const float* __restrict__ be1, const int* __restrict__ idx,
       _Float16* __restrict__ h) {
  __shared__ _Float16 As[16 * 256];
  __shared__ int sAny;
  const int tid   = threadIdx.x;
  const int lane  = tid & 31;
  const int wv    = tid >> 5;
  const int m0    = blockIdx.x * 16;
  const int ntile = blockIdx.y * 4 + wv;

  if (tid == 0) sAny = 0;
  __syncthreads();
  if (tid < 16 && idx[m0 + tid] == e) sAny = 1;
  __syncthreads();
  if (!sAny) return;   // block-uniform: EXEC stays all-1s around WMMA

  // async stage A: 512 x 16B chunks, 4 per thread (ASYNCcnt-tracked)
  {
    const char* g = (const char*)(tf16 + (size_t)m0 * C_DIM);
    unsigned base = (unsigned)(size_t)(void*)As;
#pragma unroll
    for (int j = 0; j < 4; ++j) {
      int c = tid + 128 * j;
      async_copy_b128(base + c * 16, g + c * 16);
    }
    wait_asynccnt0();
  }
  __syncthreads();

  v8f acc;
#pragma unroll
  for (int i = 0; i < 8; ++i) acc[i] = 0.f;
  const _Float16* Wb = Wt1 + (size_t)e * C_DIM * HID_D + (size_t)ntile * 16 * C_DIM;
#pragma unroll
  for (int kb = 0; kb < 8; ++kb) {
    v16h a = frag_load(&As[kb * 32], 256, lane);
    v16h b = frag_load(Wb + kb * 32, C_DIM, lane);
    acc = wmma16(a, b, acc);
  }
  int n = ntile * 16 + (lane & 15);
  float bias = be1[e * HID_D + n];
#pragma unroll
  for (int i = 0; i < 8; ++i) {
    int m   = i + ((lane >> 4) << 3);
    float v = acc[i] + bias;
    v = v > 0.f ? v : 0.f;
    h[(size_t)(m0 + m) * HID_D + n] = (_Float16)v;
  }
}

// ---------------------------------------------------------------------------
// K4: expert GEMM2: out = leaky_relu(h @ We2[e] + be2[e]) for rows routed to e.
// A tile (32KB contiguous) staged via a single TDM tensor_load_to_lds.
// ---------------------------------------------------------------------------
__global__ void __launch_bounds__(128)
k_moe2(int e, const _Float16* __restrict__ h, const _Float16* __restrict__ Wt2,
       const float* __restrict__ be2, const int* __restrict__ idx,
       float* __restrict__ out) {
  __shared__ _Float16 As[16 * 1024];
  __shared__ int sIdx[16];
  __shared__ int sAny;
  const int tid   = threadIdx.x;
  const int lane  = tid & 31;
  const int wv    = tid >> 5;
  const int m0    = blockIdx.x * 16;
  const int ntile = blockIdx.y * 4 + wv;

  if (tid == 0) sAny = 0;
  __syncthreads();
  if (tid < 16) {
    int v = idx[m0 + tid];
    sIdx[tid] = v;
    if (v == e) sAny = 1;
  }
  __syncthreads();
  if (!sAny) return;

  // TDM: one descriptor copies the full 32KB A tile (4096 x 8B) to LDS.
  if (wv == 0) {
    tdm_load_1d((unsigned)(size_t)(void*)As, h + (size_t)m0 * HID_D, 4096u);
    __builtin_amdgcn_s_wait_tensorcnt(0);
  }
  __syncthreads();

  v8f acc;
#pragma unroll
  for (int i = 0; i < 8; ++i) acc[i] = 0.f;
  const _Float16* Wb = Wt2 + (size_t)e * C_DIM * HID_D + (size_t)ntile * 16 * HID_D;
  for (int kb = 0; kb < 32; ++kb) {
    v16h a = frag_load(&As[kb * 32], 1024, lane);
    v16h b = frag_load(Wb + kb * 32, HID_D, lane);
    acc = wmma16(a, b, acc);
  }
  int n = ntile * 16 + (lane & 15);
  float bias = be2[e * C_DIM + n];
#pragma unroll
  for (int i = 0; i < 8; ++i) {
    int m = i + ((lane >> 4) << 3);
    if (sIdx[m] == e) {
      float v = acc[i] + bias;
      out[(size_t)(m0 + m) * C_DIM + n] = v >= 0.f ? v : 0.01f * v;
    }
  }
}

// ---------------------------------------------------------------------------
extern "C" void kernel_launch(void* const* d_in, const int* in_sizes, int n_in,
                              void* d_out, int out_size, void* d_ws, size_t ws_size,
                              hipStream_t stream) {
  const float* x     = (const float*)d_in[0];
  const float* w_cd  = (const float*)d_in[1];
  const float* b_cd  = (const float*)d_in[2];
  const float* w_hd  = (const float*)d_in[3];
  const float* b_hd  = (const float*)d_in[4];
  const float* w_vd  = (const float*)d_in[5];
  const float* b_vd  = (const float*)d_in[6];
  const float* w_ad  = (const float*)d_in[7];
  const float* b_ad  = (const float*)d_in[8];
  const float* w_std = (const float*)d_in[9];
  const float* b_std = (const float*)d_in[10];
  const float* Wg    = (const float*)d_in[11];
  const float* bg    = (const float*)d_in[12];
  const float* Wn    = (const float*)d_in[13];
  const float* bn    = (const float*)d_in[14];
  const float* We1   = (const float*)d_in[15];
  const float* be1   = (const float*)d_in[16];
  const float* We2   = (const float*)d_in[17];
  const float* be2   = (const float*)d_in[18];
  const float* noise = (const float*)d_in[19];
  float* out = (float*)d_out;

  char* ws = (char*)d_ws;
  _Float16* Wt   = (_Float16*)(ws + 0);
  float*    bf   = (float*)   (ws + 1179648);
  float*    t    = (float*)   (ws + 1180672);
  _Float16* tf16 = (_Float16*)(ws + 9569280);
  int*      idx  = (int*)     (ws + 13763584);
  _Float16* Wt1  = (_Float16*)(ws + 13796352);
  _Float16* Wt2  = (_Float16*)(ws + 16417792);
  _Float16* hbuf = (_Float16*)(ws + 19039232);

  k_reparam<<<dim3(2304), dim3(256), 0, stream>>>(w_cd, b_cd, w_hd, b_hd, w_vd, b_vd,
                                                  w_ad, b_ad, w_std, b_std, Wt, bf);
  k_transpose_experts<<<dim3(5120), dim3(256), 0, stream>>>(We1, We2, Wt1, Wt2);
  k_conv<<<dim3(512), dim3(256), 0, stream>>>(x, Wt, bf, t, tf16);
  k_router<<<dim3(32), dim3(256), 0, stream>>>(t, Wg, bg, Wn, bn, noise, idx);
  for (int e = 0; e < E_NUM; ++e) {
    k_moe1<<<dim3(512, 16), dim3(128), 0, stream>>>(e, tf16, Wt1, be1, idx, hbuf);
    k_moe2<<<dim3(512, 4),  dim3(128), 0, stream>>>(e, hbuf, Wt2, be2, idx, out);
  }
}
